// QuantFusedMLP_11261404250116
// MI455X (gfx1250) — compile-verified, matrix operand order checked by reading, not run on hardware
//
#include <hip/hip_runtime.h>
#include <hip/hip_fp16.h>
#include <stdint.h>

typedef __attribute__((ext_vector_type(16))) _Float16 v16h;
typedef __attribute__((ext_vector_type(8)))  _Float16 v8h;
typedef __attribute__((ext_vector_type(4)))  _Float16 v4h;
typedef __attribute__((ext_vector_type(2)))  _Float16 v2h;
typedef __attribute__((ext_vector_type(8)))  float    v8f;

static constexpr int M_TOK = 4096;    // B*S = 2*2048
static constexpr int D_IN  = 4096;    // hidden dim
static constexpr int I_FF  = 11008;   // intermediate dim
static constexpr int BM = 128, BN = 64, BK = 64;
static constexpr int LDK = BK + 8;    // padded row stride (144 B, 16B-multiple)
static constexpr int THREADS = 256;   // 8 waves (wave32)

__device__ __forceinline__ v8f wmma_f16(v16h a, v16h b, v8f c) {
  return __builtin_amdgcn_wmma_f32_16x16x32_f16(false, a, false, b, (short)0, c,
                                                false, false);
}

// A/B fragment per CDNA5 16-bit layout: lanes 0-15 hold K = 0..7,16..23;
// lanes 16-31 hold K = 8..15,24..31 (two ds_load_b128 per fragment).
__device__ __forceinline__ v16h ld_frag(const _Float16* rowp, int hi8) {
  v8h lo = *(const v8h*)(rowp + hi8);
  v8h hi = *(const v8h*)(rowp + 16 + hi8);
  v16h r;
#pragma unroll
  for (int i = 0; i < 8; ++i) { r[i] = lo[i]; r[i + 8] = hi[i]; }
  return r;
}

// gfx1250 async DMA: global -> LDS, 16 bytes per lane, tracked by ASYNCcnt.
__device__ __forceinline__ void async_copy_b128(const _Float16* gp, uint32_t lds_addr) {
  asm volatile("global_load_async_to_lds_b128 %0, %1, off"
               :: "v"(lds_addr), "v"((unsigned long long)(uintptr_t)gp)
               : "memory");
}

__device__ __forceinline__ void wait_asynccnt0() {
#if __has_builtin(__builtin_amdgcn_s_wait_asynccnt)
  __builtin_amdgcn_s_wait_asynccnt(0);
#else
  asm volatile("s_wait_asynccnt 0" ::: "memory");
#endif
}

// ---------------------------------------------------------------- x fp32 -> f16
__global__ __launch_bounds__(256)
void cvt_f32_f16(const float* __restrict__ x, _Float16* __restrict__ y) {
  const int i = (blockIdx.x * 256 + threadIdx.x) * 4;
  const float4 v = *(const float4*)(x + i);
  v4h h;
  h[0] = (_Float16)v.x; h[1] = (_Float16)v.y;
  h[2] = (_Float16)v.z; h[3] = (_Float16)v.w;
  *(v4h*)(y + i) = h;
}

// ------------------------------------------------- fused gate+up int4 GEMM + silu
__global__ __launch_bounds__(THREADS)
void gateup_kernel(const _Float16* __restrict__ x16,
                   const uint32_t* __restrict__ gqw, const float* __restrict__ gsc,
                   const uint32_t* __restrict__ gqz,
                   const uint32_t* __restrict__ uqw, const float* __restrict__ usc,
                   const uint32_t* __restrict__ uqz,
                   _Float16* __restrict__ hbuf) {
  __shared__ __align__(16) _Float16 As[2][BM][LDK];   // x tile  [M][K], double-buffered
  __shared__ __align__(16) _Float16 Bg[2][BN][LDK];   // gate    [N][K]
  __shared__ __align__(16) _Float16 Bu[2][BN][LDK];   // up      [N][K]

  const int tid  = threadIdx.x;
  const int lane = tid & 31;
  const int wave = tid >> 5;
  const int wm   = wave & 3;            // M sub-block (32 rows)
  const int wn   = wave >> 2;           // N sub-block (32 cols)
  const int hl   = lane & 15;
  const int hi8  = (lane >> 4) << 3;

  const int mBlock = blockIdx.y * BM;
  const int nBlock = blockIdx.x * BN;

  const int j   = tid >> 5;             // packed word column 0..7 (8 words = 64 n)
  const int kp  = tid & 31;             // k-pair index: rows 2kp, 2kp+1
  const int ldq = I_FF / 8;             // 1376 int32 per K row
  const int nb8 = nBlock / 8;

  const int kShift[8] = {0, 16, 4, 20, 8, 24, 12, 28};  // AWQ nibble order

  float sg[8], su[8];
  int   zg[8], zu[8];
  int curGroup = 0;

  v8f accG[2][2], accU[2][2];
#pragma unroll
  for (int a = 0; a < 2; ++a)
#pragma unroll
    for (int b = 0; b < 2; ++b)
#pragma unroll
      for (int r = 0; r < 8; ++r) { accG[a][b][r] = 0.0f; accU[a][b][r] = 0.0f; }

  constexpr int NT = D_IN / BK;         // 64 K-tiles

  // -------- prologue: stage tile 0 into buffer 0 --------
  {
    const uint32_t zwg = gqz[nb8 + j];
    const uint32_t zwu = uqz[nb8 + j];
#pragma unroll
    for (int c = 0; c < 8; ++c) {
      zg[c] = (int)((zwg >> kShift[c]) & 15u);
      zu[c] = (int)((zwu >> kShift[c]) & 15u);
      sg[c] = gsc[nBlock + j * 8 + c];
      su[c] = usc[nBlock + j * 8 + c];
    }
#pragma unroll
    for (int cc = 0; cc < 4; ++cc) {
      const int ch = tid + cc * 256;
      const int r = ch >> 3, o = (ch & 7) * 8;
      async_copy_b128(x16 + (size_t)(mBlock + r) * D_IN + o,
                      (uint32_t)(uintptr_t)(&As[0][r][o]));
    }
    const size_t wrow = (size_t)(2 * kp) * ldq + nb8 + j;
    const uint32_t wg0 = gqw[wrow], wg1 = gqw[wrow + ldq];
    const uint32_t wu0 = uqw[wrow], wu1 = uqw[wrow + ldq];
#pragma unroll
    for (int c = 0; c < 8; ++c) {
      const int sh = kShift[c];
      v2h pg, pu;
      pg[0] = (_Float16)((float)((int)((wg0 >> sh) & 15u) - zg[c]) * sg[c]);
      pg[1] = (_Float16)((float)((int)((wg1 >> sh) & 15u) - zg[c]) * sg[c]);
      pu[0] = (_Float16)((float)((int)((wu0 >> sh) & 15u) - zu[c]) * su[c]);
      pu[1] = (_Float16)((float)((int)((wu1 >> sh) & 15u) - zu[c]) * su[c]);
      *(v2h*)(&Bg[0][j * 8 + c][2 * kp]) = pg;
      *(v2h*)(&Bu[0][j * 8 + c][2 * kp]) = pu;
    }
    wait_asynccnt0();
    __syncthreads();
  }

  // -------- pipelined main loop: one barrier per K-tile --------
  for (int t = 0; t < NT; ++t) {
    const int buf = t & 1;
    const int nxt = buf ^ 1;
    const int nk0 = (t + 1) * BK;
    const bool hasNext = (t + 1) < NT;

    uint32_t wg0 = 0, wg1 = 0, wu0 = 0, wu1 = 0;
    if (hasNext) {
      // async DMA of next A panel runs underneath this tile's WMMAs
#pragma unroll
      for (int cc = 0; cc < 4; ++cc) {
        const int ch = tid + cc * 256;
        const int r = ch >> 3, o = (ch & 7) * 8;
        async_copy_b128(x16 + (size_t)(mBlock + r) * D_IN + nk0 + o,
                        (uint32_t)(uintptr_t)(&As[nxt][r][o]));
      }
      const int g = nk0 >> 7;
      if (g != curGroup) {
        curGroup = g;
        const uint32_t zwg = gqz[(size_t)g * ldq + nb8 + j];
        const uint32_t zwu = uqz[(size_t)g * ldq + nb8 + j];
#pragma unroll
        for (int c = 0; c < 8; ++c) {
          zg[c] = (int)((zwg >> kShift[c]) & 15u);
          zu[c] = (int)((zwu >> kShift[c]) & 15u);
          sg[c] = gsc[(size_t)g * I_FF + nBlock + j * 8 + c];
          su[c] = usc[(size_t)g * I_FF + nBlock + j * 8 + c];
        }
      }
      const size_t wrow = (size_t)(nk0 + 2 * kp) * ldq + nb8 + j;
      wg0 = gqw[wrow];
      wg1 = gqw[wrow + ldq];
      wu0 = uqw[wrow];
      wu1 = uqw[wrow + ldq];
      if (nk0 + BK < D_IN) {
        __builtin_prefetch(gqw + wrow + (size_t)BK * ldq, 0, 1);
        __builtin_prefetch(uqw + wrow + (size_t)BK * ldq, 0, 1);
      }
    }

    // 16 WMMAs on the current buffer
#pragma unroll
    for (int ks = 0; ks < 2; ++ks) {
      const int co = ks * 32;
      const v16h a0 = ld_frag(&As[buf][wm * 32 + hl][co],      hi8);
      const v16h a1 = ld_frag(&As[buf][wm * 32 + 16 + hl][co], hi8);
#pragma unroll
      for (int ns = 0; ns < 2; ++ns) {
        const v16h bgf = ld_frag(&Bg[buf][wn * 32 + ns * 16 + hl][co], hi8);
        accG[0][ns] = wmma_f16(a0, bgf, accG[0][ns]);
        accG[1][ns] = wmma_f16(a1, bgf, accG[1][ns]);
        const v16h buv = ld_frag(&Bu[buf][wn * 32 + ns * 16 + hl][co], hi8);
        accU[0][ns] = wmma_f16(a0, buv, accU[0][ns]);
        accU[1][ns] = wmma_f16(a1, buv, accU[1][ns]);
      }
    }

    if (hasNext) {
      // dequant next tile into the other buffer (no hazard with readers)
#pragma unroll
      for (int c = 0; c < 8; ++c) {
        const int sh = kShift[c];
        v2h pg, pu;
        pg[0] = (_Float16)((float)((int)((wg0 >> sh) & 15u) - zg[c]) * sg[c]);
        pg[1] = (_Float16)((float)((int)((wg1 >> sh) & 15u) - zg[c]) * sg[c]);
        pu[0] = (_Float16)((float)((int)((wu0 >> sh) & 15u) - zu[c]) * su[c]);
        pu[1] = (_Float16)((float)((int)((wu1 >> sh) & 15u) - zu[c]) * su[c]);
        *(v2h*)(&Bg[nxt][j * 8 + c][2 * kp]) = pg;
        *(v2h*)(&Bu[nxt][j * 8 + c][2 * kp]) = pu;
      }
    }

    wait_asynccnt0();
    __syncthreads();
  }

  // epilogue: h = silu(gate) * up -> f16
#pragma unroll
  for (int ms = 0; ms < 2; ++ms)
#pragma unroll
    for (int ns = 0; ns < 2; ++ns) {
      const int rowBase = mBlock + wm * 32 + ms * 16 + ((lane >> 4) << 3);
      const int col     = nBlock + wn * 32 + ns * 16 + hl;
#pragma unroll
      for (int r = 0; r < 8; ++r) {
        const float gv = accG[ms][ns][r];
        const float uv = accU[ms][ns][r];
        const float hv = gv / (1.0f + __expf(-gv)) * uv;
        hbuf[(size_t)(rowBase + r) * I_FF + col] = (_Float16)hv;
      }
    }
}

// ------------------------------------------------------------- down int4 GEMM
__global__ __launch_bounds__(THREADS)
void down_kernel(const _Float16* __restrict__ hbuf,
                 const uint32_t* __restrict__ dqw, const float* __restrict__ dsc,
                 const uint32_t* __restrict__ dqz,
                 float* __restrict__ out) {
  __shared__ __align__(16) _Float16 As[2][BM][LDK];
  __shared__ __align__(16) _Float16 Bd[2][BN][LDK];

  const int tid  = threadIdx.x;
  const int lane = tid & 31;
  const int wave = tid >> 5;
  const int wm   = wave & 3;
  const int wn   = wave >> 2;
  const int hl   = lane & 15;
  const int hi8  = (lane >> 4) << 3;

  const int mBlock = blockIdx.y * BM;
  const int nBlock = blockIdx.x * BN;

  const int j   = tid >> 5;
  const int kp  = tid & 31;
  const int ldq = D_IN / 8;   // 512
  const int nb8 = nBlock / 8;

  const int kShift[8] = {0, 16, 4, 20, 8, 24, 12, 28};

  float sd[8];
  int   zd[8];
  int curGroup = 0;

  v8f acc[2][2];
#pragma unroll
  for (int a = 0; a < 2; ++a)
#pragma unroll
    for (int b = 0; b < 2; ++b)
#pragma unroll
      for (int r = 0; r < 8; ++r) acc[a][b][r] = 0.0f;

  constexpr int NT = I_FF / BK;   // 172

  // -------- prologue: stage tile 0 --------
  {
    const uint32_t zw = dqz[nb8 + j];
#pragma unroll
    for (int c = 0; c < 8; ++c) {
      zd[c] = (int)((zw >> kShift[c]) & 15u);
      sd[c] = dsc[nBlock + j * 8 + c];
    }
#pragma unroll
    for (int cc = 0; cc < 4; ++cc) {
      const int ch = tid + cc * 256;
      const int r = ch >> 3, o = (ch & 7) * 8;
      async_copy_b128(hbuf + (size_t)(mBlock + r) * I_FF + o,
                      (uint32_t)(uintptr_t)(&As[0][r][o]));
    }
    const size_t wrow = (size_t)(2 * kp) * ldq + nb8 + j;
    const uint32_t w0 = dqw[wrow], w1 = dqw[wrow + ldq];
#pragma unroll
    for (int c = 0; c < 8; ++c) {
      const int sh = kShift[c];
      v2h p;
      p[0] = (_Float16)((float)((int)((w0 >> sh) & 15u) - zd[c]) * sd[c]);
      p[1] = (_Float16)((float)((int)((w1 >> sh) & 15u) - zd[c]) * sd[c]);
      *(v2h*)(&Bd[0][j * 8 + c][2 * kp]) = p;
    }
    wait_asynccnt0();
    __syncthreads();
  }

  // -------- pipelined main loop --------
  for (int t = 0; t < NT; ++t) {
    const int buf = t & 1;
    const int nxt = buf ^ 1;
    const int nk0 = (t + 1) * BK;
    const bool hasNext = (t + 1) < NT;

    uint32_t w0 = 0, w1 = 0;
    if (hasNext) {
#pragma unroll
      for (int cc = 0; cc < 4; ++cc) {
        const int ch = tid + cc * 256;
        const int r = ch >> 3, o = (ch & 7) * 8;
        async_copy_b128(hbuf + (size_t)(mBlock + r) * I_FF + nk0 + o,
                        (uint32_t)(uintptr_t)(&As[nxt][r][o]));
      }
      const int g = nk0 >> 7;
      if (g != curGroup) {
        curGroup = g;
        const uint32_t zw = dqz[(size_t)g * ldq + nb8 + j];
#pragma unroll
        for (int c = 0; c < 8; ++c) {
          zd[c] = (int)((zw >> kShift[c]) & 15u);
          sd[c] = dsc[(size_t)g * D_IN + nBlock + j * 8 + c];
        }
      }
      const size_t wrow = (size_t)(nk0 + 2 * kp) * ldq + nb8 + j;
      w0 = dqw[wrow];
      w1 = dqw[wrow + ldq];
      if (nk0 + BK < I_FF)
        __builtin_prefetch(dqw + wrow + (size_t)BK * ldq, 0, 1);
    }

#pragma unroll
    for (int ks = 0; ks < 2; ++ks) {
      const int co = ks * 32;
      const v16h a0 = ld_frag(&As[buf][wm * 32 + hl][co],      hi8);
      const v16h a1 = ld_frag(&As[buf][wm * 32 + 16 + hl][co], hi8);
#pragma unroll
      for (int ns = 0; ns < 2; ++ns) {
        const v16h bf = ld_frag(&Bd[buf][wn * 32 + ns * 16 + hl][co], hi8);
        acc[0][ns] = wmma_f16(a0, bf, acc[0][ns]);
        acc[1][ns] = wmma_f16(a1, bf, acc[1][ns]);
      }
    }

    if (hasNext) {
#pragma unroll
      for (int c = 0; c < 8; ++c) {
        const int sh = kShift[c];
        v2h p;
        p[0] = (_Float16)((float)((int)((w0 >> sh) & 15u) - zd[c]) * sd[c]);
        p[1] = (_Float16)((float)((int)((w1 >> sh) & 15u) - zd[c]) * sd[c]);
        *(v2h*)(&Bd[nxt][j * 8 + c][2 * kp]) = p;
      }
    }

    wait_asynccnt0();
    __syncthreads();
  }

#pragma unroll
  for (int ms = 0; ms < 2; ++ms)
#pragma unroll
    for (int ns = 0; ns < 2; ++ns) {
      const int rowBase = mBlock + wm * 32 + ms * 16 + ((lane >> 4) << 3);
      const int col     = nBlock + wn * 32 + ns * 16 + hl;
#pragma unroll
      for (int r = 0; r < 8; ++r)
        out[(size_t)(rowBase + r) * D_IN + col] = acc[ms][ns][r];
    }
}

// ------------------------------------------------------------------ launcher
extern "C" void kernel_launch(void* const* d_in, const int* in_sizes, int n_in,
                              void* d_out, int out_size, void* d_ws, size_t ws_size,
                              hipStream_t stream) {
  const float*    x   = (const float*)   d_in[0];
  const uint32_t* gqw = (const uint32_t*)d_in[1];
  const float*    gsc = (const float*)   d_in[2];
  const uint32_t* gqz = (const uint32_t*)d_in[3];
  const uint32_t* uqw = (const uint32_t*)d_in[4];
  const float*    usc = (const float*)   d_in[5];
  const uint32_t* uqz = (const uint32_t*)d_in[6];
  const uint32_t* dqw = (const uint32_t*)d_in[7];
  const float*    dsc = (const float*)   d_in[8];
  const uint32_t* dqz = (const uint32_t*)d_in[9];
  float*          out = (float*)d_out;

  _Float16* x16  = (_Float16*)d_ws;                                      // 32 MB
  _Float16* hbuf = (_Float16*)((char*)d_ws +
                               (size_t)M_TOK * D_IN * sizeof(_Float16)); // 90 MB

  const int nconv = M_TOK * D_IN;
  cvt_f32_f16<<<nconv / (256 * 4), 256, 0, stream>>>(x, x16);

  gateup_kernel<<<dim3(I_FF / BN, M_TOK / BM), THREADS, 0, stream>>>(
      x16, gqw, gsc, gqz, uqw, usc, uqz, hbuf);

  down_kernel<<<dim3(D_IN / BN, M_TOK / BM), THREADS, 0, stream>>>(
      hbuf, dqw, dsc, dqz, out);
}
